// Encoder_63814624084457
// MI455X (gfx1250) — matrix operand order repaired and checked by reading, not compiled, not run on hardware
//
#include <hip/hip_runtime.h>
#include <hip/hip_bf16.h>

typedef __attribute__((ext_vector_type(16))) __bf16 v16bf;
typedef __attribute__((ext_vector_type(8)))  float  v8f;

#define M_DIM 16384
#define N_DIM 4096
#define K_DIM 4096
#define TOPK  128

#define BM 128
#define BN 128
#define BK 32
#define LDS_STRIDE 40   // 32 + 8 pad (bf16 elems) -> 80B rows, keeps b128 alignment

// pack two fp32 into one dword of bf16 {lo, hi}
__device__ __forceinline__ unsigned pack2bf(float lo, float hi) {
#if __has_builtin(__builtin_amdgcn_cvt_pk_bf16_f32)
    auto r = __builtin_amdgcn_cvt_pk_bf16_f32(lo, hi);
    unsigned u;
    __builtin_memcpy(&u, &r, sizeof(u));
    return u;
#else
    // builtin is gated to gfx950 in this toolchain; probe the CDNA ISA op
    // directly through the assembler (single VALU op, RNE rounding).
    unsigned r;
    asm("v_cvt_pk_bf16_f32 %0, %1, %2" : "=v"(r) : "v"(lo), "v"(hi));
    return r;
#endif
}

__device__ __forceinline__ v16bf load_frag(const unsigned short* p0,
                                           const unsigned short* p1) {
    union { uint4 u[2]; v16bf v; } f;
    f.u[0] = *(const uint4*)p0;
    f.u[1] = *(const uint4*)p1;
    return f.v;
}

__global__ __launch_bounds__(256)
void encoder_gemm_bf16(const float* __restrict__ x,
                       const float* __restrict__ pre_bias,
                       const float* __restrict__ W,   // [K][N] row-major
                       const float* __restrict__ b_enc,
                       float* __restrict__ out) {
    __shared__ unsigned short Asm[2][BM][LDS_STRIDE];  // [m][k] bf16
    __shared__ unsigned short Bsm[2][BN][LDS_STRIDE];  // [n][k] bf16 (transposed)

    const int tid  = threadIdx.x;
    const int lane = tid & 31;
    const int wave = tid >> 5;           // 0..7
    const int waveM = wave >> 2;         // 0..1  -> 64 rows each
    const int waveN = wave & 3;          // 0..3  -> 32 cols each

    const int n0 = blockIdx.x * BN;
    const int m0 = blockIdx.y * BM;

    v8f acc[4][2] = {};

    const int lm  = lane & 15;           // M / N within 16x16 frag
    const int kbA = (lane >> 4) * 8;     // A frag: K base (0 or 8)
    const int kbB = (lane >> 4) * 8;     // B frag: K base (0 or 8)

    // staging registers (next tile)
    float4 xa[4];        // A: 4 float4 (bias already subtracted)
    float4 wb[2][2];     // B: 2 units x (row k, row k+1) float4

    auto globalLoad = [&](int k0) {
        #pragma unroll
        for (int i = 0; i < 4; ++i) {
            int f   = tid + 256 * i;          // 0..1023 float4 slots
            int row = f >> 3;                 // 0..127
            int c4  = f & 7;                  // 0..7
            float4 xv = *(const float4*)(x + (size_t)(m0 + row) * K_DIM + k0 + c4 * 4);
            float4 pb = *(const float4*)(pre_bias + k0 + c4 * 4);
            xv.x -= pb.x; xv.y -= pb.y; xv.z -= pb.z; xv.w -= pb.w;
            xa[i] = xv;
        }
        #pragma unroll
        for (int u = 0; u < 2; ++u) {
            int f  = tid + 256 * u;           // 0..511 units (2K x 4N each)
            int kk = (f >> 5) * 2;            // 0,2,..,30
            int c4 = f & 31;                  // 0..31 along N
            wb[u][0] = *(const float4*)(W + (size_t)(k0 + kk)     * N_DIM + n0 + c4 * 4);
            wb[u][1] = *(const float4*)(W + (size_t)(k0 + kk + 1) * N_DIM + n0 + c4 * 4);
        }
    };

    auto ldsStore = [&](int buf) {
        #pragma unroll
        for (int i = 0; i < 4; ++i) {
            int f   = tid + 256 * i;
            int row = f >> 3;
            int c4  = f & 7;
            uint2 d;
            d.x = pack2bf(xa[i].x, xa[i].y);
            d.y = pack2bf(xa[i].z, xa[i].w);
            *(uint2*)&Asm[buf][row][c4 * 4] = d;
        }
        #pragma unroll
        for (int u = 0; u < 2; ++u) {
            int f  = tid + 256 * u;
            int kk = (f >> 5) * 2;
            int c4 = f & 31;
            const float* r0 = reinterpret_cast<const float*>(&wb[u][0]);
            const float* r1 = reinterpret_cast<const float*>(&wb[u][1]);
            #pragma unroll
            for (int j = 0; j < 4; ++j)
                *(unsigned*)&Bsm[buf][c4 * 4 + j][kk] = pack2bf(r0[j], r1[j]);
        }
    };

    globalLoad(0);
    ldsStore(0);
    __syncthreads();

    int buf = 0;
    for (int k0 = 0; k0 < K_DIM; k0 += BK, buf ^= 1) {
        const int next = k0 + BK;
        if (next < K_DIM) globalLoad(next);   // issue early: hide behind WMMA

        v16bf afrag[4];
        #pragma unroll
        for (int mi = 0; mi < 4; ++mi) {
            int mloc = waveM * 64 + mi * 16 + lm;
            afrag[mi] = load_frag(&Asm[buf][mloc][kbA], &Asm[buf][mloc][kbA + 16]);
        }
        v16bf bfrag[2];
        #pragma unroll
        for (int ni = 0; ni < 2; ++ni) {
            int nloc = waveN * 32 + ni * 16 + lm;
            bfrag[ni] = load_frag(&Bsm[buf][nloc][kbB], &Bsm[buf][nloc][kbB + 16]);
        }
        #pragma unroll
        for (int mi = 0; mi < 4; ++mi)
            #pragma unroll
            for (int ni = 0; ni < 2; ++ni)
                acc[mi][ni] = __builtin_amdgcn_wmma_f32_16x16x32_bf16(
                    false, afrag[mi], false, bfrag[ni],
                    (short)0, acc[mi][ni], false, false);

        if (next < K_DIM) ldsStore(buf ^ 1);  // fill other buffer
        __syncthreads();
    }

    // ---- epilogue: + b_enc, ReLU, store fp32 ----
    const int rowsel = (lane >> 4) * 8;      // lanes 16-31 hold M+8..M+15
    #pragma unroll
    for (int ni = 0; ni < 2; ++ni) {
        const int col = n0 + waveN * 32 + ni * 16 + lm;
        const float bv = b_enc[col];
        #pragma unroll
        for (int mi = 0; mi < 4; ++mi) {
            const int rbase = m0 + waveM * 64 + mi * 16 + rowsel;
            #pragma unroll
            for (int r = 0; r < 8; ++r) {
                float v = acc[mi][ni][r] + bv;
                v = fmaxf(v, 0.0f);
                out[(size_t)(rbase + r) * N_DIM + col] = v;
            }
        }
    }
}

// Exact top-K threshold per row via 4-pass MSD radix select on float bits
// (all values are >= 0 after ReLU, so uint order == float order).
// Row is staged into LDS with CDNA5 async global->LDS copies (ASYNCcnt).
__global__ __launch_bounds__(256)
void topk_threshold(float* __restrict__ out) {
    __shared__ unsigned vals[N_DIM];
    __shared__ unsigned hist[256];
    __shared__ unsigned s_sel, s_k;

    const int t = threadIdx.x;
    const size_t base = (size_t)blockIdx.x * N_DIM;

    // async copy 16KB row directly into LDS (no VGPR round-trip)
    #pragma unroll
    for (int i = 0; i < 4; ++i) {
        int idx = (t + 256 * i) * 4;                         // float index, 16B chunks
        unsigned ldsAddr = (unsigned)(size_t)&vals[idx];     // LDS_ADDR = flat[31:0]
        const float* g = out + base + idx;
        asm volatile("global_load_async_to_lds_b128 %0, %1, off"
                     :: "v"(ldsAddr), "v"(g) : "memory");
    }
    asm volatile("s_wait_asynccnt 0" ::: "memory");
    __syncthreads();

    unsigned prefix = 0;
    unsigned k = TOPK;
    #pragma unroll
    for (int p = 0; p < 4; ++p) {
        const int shift = 24 - 8 * p;
        hist[t] = 0;
        __syncthreads();
        const unsigned maskHigh = (p == 0) ? 0u : (0xFFFFFFFFu << (shift + 8));
        #pragma unroll
        for (int i = 0; i < 16; ++i) {
            unsigned v = vals[t + 256 * i];
            if ((v & maskHigh) == (prefix & maskHigh))
                atomicAdd(&hist[(v >> shift) & 0xFFu], 1u);
        }
        __syncthreads();
        if (t == 0) {
            unsigned cum = 0, sel = 0, above = 0;
            for (int b = 255; b >= 0; --b) {
                unsigned h = hist[b];
                if (cum + h >= k) { sel = (unsigned)b; above = cum; break; }
                cum += h;
            }
            s_sel = sel;
            s_k   = k - above;
        }
        __syncthreads();
        prefix |= (s_sel << shift);
        k = s_k;
        __syncthreads();
    }

    const float thr = __uint_as_float(prefix);
    #pragma unroll
    for (int i = 0; i < 16; ++i) {
        int idx = t + 256 * i;
        float v = __uint_as_float(vals[idx]);
        out[base + idx] = (v >= thr) ? v : 0.0f;
    }
}

extern "C" void kernel_launch(void* const* d_in, const int* in_sizes, int n_in,
                              void* d_out, int out_size, void* d_ws, size_t ws_size,
                              hipStream_t stream) {
    const float* x        = (const float*)d_in[0];
    const float* pre_bias = (const float*)d_in[1];
    const float* W_enc    = (const float*)d_in[2];
    const float* b_enc    = (const float*)d_in[3];
    float* out = (float*)d_out;

    dim3 grid(N_DIM / BN, M_DIM / BM);   // (32, 128)
    encoder_gemm_bf16<<<grid, 256, 0, stream>>>(x, pre_bias, W_enc, b_enc, out);
    topk_threshold<<<M_DIM, 256, 0, stream>>>(out);
}
